// NtXentLoss_21345987461325
// MI455X (gfx1250) — compile-verified, matrix operand order checked by reading, not run on hardware
//
#include <hip/hip_runtime.h>
#include <hip/hip_bf16.h>

// NT-Xent loss, MI455X (gfx1250, wave32).
// Split-bf16 WMMA GEMM (hi/lo decomposition ~ f32 accuracy) + fused
// symmetrized exp-rowsum reduction over upper-triangular tiles.
// v3: tensor_load_to_lds uses the 6-arg builtin form (this toolchain).

typedef __attribute__((ext_vector_type(16))) __bf16 v16bf;
typedef __attribute__((ext_vector_type(8)))  float  v8f;
typedef __attribute__((ext_vector_type(4)))  unsigned int v4u;
typedef __attribute__((ext_vector_type(8)))  int    v8i;
typedef __attribute__((ext_vector_type(4)))  int    v4i;

#define B_SZ   4096
#define D_SZ   1024
#define TILE   128
#define KSTEP  32
#define LDST   40      // LDS row stride in ushort: 80B -> conflict-free & 16B aligned
#define INV_T  10.0f   // 1 / temperature
#define EPS_N  1e-8f

#if defined(__has_builtin)
#if __has_builtin(__builtin_amdgcn_tensor_load_to_lds) && __has_builtin(__builtin_amdgcn_s_wait_tensorcnt)
#define HAVE_TDM 1
#endif
#endif

__device__ __forceinline__ unsigned short f32_to_bf16_rne(float x) {
  unsigned u = __float_as_uint(x);
  unsigned r = u + 0x7FFFu + ((u >> 16) & 1u);   // round-to-nearest-even
  return (unsigned short)(r >> 16);
}
__device__ __forceinline__ float bf16u_to_f32(unsigned short h) {
  return __uint_as_float(((unsigned)h) << 16);
}

union FragU { v16bf v; uint4 q[2]; };

__device__ __forceinline__ v16bf load_frag(const unsigned short* lds, int row, int halfk) {
  // Per-lane A/B fragment, ISA 16-bit 16x32 layout:
  //   elements 0..7  <-> K = halfk*8 + {0..7}
  //   elements 8..15 <-> K = 16 + halfk*8 + {0..7}
  FragU f;
  const unsigned short* p = lds + row * LDST;
  f.q[0] = *(const uint4*)(p + halfk * 8);
  f.q[1] = *(const uint4*)(p + 16 + halfk * 8);
  return f.v;
}

#define WMMA_BF16(A, Bm, C) \
  __builtin_amdgcn_wmma_f32_16x16x32_bf16(false, (A), false, (Bm), (short)0, (C), false, false)

// 24 WMMAs per K-step: 2x4 subtiles x (hi*hi + hi*lo + lo*hi)
__device__ __forceinline__ void compute_step(const unsigned short* sAhi, const unsigned short* sAlo,
                                             const unsigned short* sBhi, const unsigned short* sBlo,
                                             v8f acc[2][4], int wrBase, int wcBase, int l16, int half) {
  v16bf ah[2], al[2], bh[4], bl[4];
  for (int sr = 0; sr < 2; ++sr) {
    const int row = wrBase + sr * 16 + l16;
    ah[sr] = load_frag(sAhi, row, half);
    al[sr] = load_frag(sAlo, row, half);
  }
  for (int sc = 0; sc < 4; ++sc) {
    const int col = wcBase + sc * 16 + l16;
    bh[sc] = load_frag(sBhi, col, half);
    bl[sc] = load_frag(sBlo, col, half);
  }
  for (int sr = 0; sr < 2; ++sr)
    for (int sc = 0; sc < 4; ++sc) {
      acc[sr][sc] = WMMA_BF16(ah[sr], bh[sc], acc[sr][sc]);
      acc[sr][sc] = WMMA_BF16(ah[sr], bl[sc], acc[sr][sc]);
      acc[sr][sc] = WMMA_BF16(al[sr], bh[sc], acc[sr][sc]);
    }
}

// cos -> exp(cos/T), symmetrized row/col reductions, diag capture
__device__ __forceinline__ void nt_epilogue(v8f acc[2][4], const float* lnS, const float* rnS,
                                            float* rowsum, float* diag,
                                            bool diagTile, int biBase, int bjBase,
                                            int wrBase, int wcBase, int l16, int half) {
  float rp[2][8];
  float cs[4] = {0.f, 0.f, 0.f, 0.f};
  for (int sr = 0; sr < 2; ++sr)
    for (int r = 0; r < 8; ++r) rp[sr][r] = 0.f;

  for (int sr = 0; sr < 2; ++sr)
    for (int sc = 0; sc < 4; ++sc) {
      const int lj = wcBase + sc * 16 + l16;
      const float rnj = rnS[lj];
      for (int r = 0; r < 8; ++r) {
        const int li = wrBase + sr * 16 + r + 8 * half;   // C layout: M = r + 8*half
        const float dot = acc[sr][sc][r];
        const float s = dot / fmaxf(lnS[li] * rnj, EPS_N) * INV_T;
        const float e = __expf(s);                        // |s| <= ~10: always safe
        float erow = e, ecol = e;
        if (diagTile) {
          if (lj < li)  erow = 0.f;                       // lower triangle unused
          if (lj <= li) ecol = 0.f;                       // diag counted once (row side)
          if (li == lj) diag[biBase + li] = s;            // written exactly once
        }
        rp[sr][r] += erow;
        cs[sc]    += ecol;
      }
    }

  for (int sc = 0; sc < 4; ++sc) {                        // column sums
    float v = cs[sc];
    v += __shfl_xor(v, 16, 32);
    if (half == 0) atomicAdd(&rowsum[bjBase + wcBase + sc * 16 + l16], v);
  }
  for (int sr = 0; sr < 2; ++sr)                          // row sums
    for (int r = 0; r < 8; ++r) {
      float v = rp[sr][r];
      v += __shfl_xor(v, 1, 32);
      v += __shfl_xor(v, 2, 32);
      v += __shfl_xor(v, 4, 32);
      v += __shfl_xor(v, 8, 32);
      if (l16 == 0) atomicAdd(&rowsum[biBase + wrBase + sr * 16 + r + 8 * half], v);
    }
}

#ifdef HAVE_TDM
// ---- Tensor Data Mover: one 128x32 bf16 tile, global -> padded LDS ----
__device__ __forceinline__ void tdm_tile_load(const unsigned short* gbase, long long elemOff,
                                              const unsigned short* ldsPtr, v8i g1) {
  const unsigned long long ga = (unsigned long long)gbase + (unsigned long long)elemOff * 2ull;
  v4u g0;
  g0.x = 1u;                                              // count=1 (valid user D#)
  g0.y = (unsigned)(unsigned long long)(size_t)ldsPtr;    // lds_addr (low 32b of generic)
  g0.z = (unsigned)ga;                                    // global_addr[31:0]
  g0.w = ((unsigned)(ga >> 32) & 0x01FFFFFFu) | (2u << 30); // global_addr[56:32] | type=2
  __builtin_amdgcn_tensor_load_to_lds(g0, g1, (v4i)0, (v4i)0, (v8i)0, 0);
}

__device__ __forceinline__ v8i tdm_group1(void) {
  // data_size=2B(code 1); pad_enable; pad_interval=16 DWORDs(code 3) -> pad after each
  // 64B tile row; pad_amount=4 DWORDs(code 3) -> LDS row stride 40 ushorts == LDST.
  // tensor_dim0/1 oversized (tiles never reach OOB); tile = 32 x 128;
  // tensor_dim0_stride = 1024 elements.
  v8i g1 = { (1 << 16) | (1 << 20) | (3 << 22) | (3 << 25),
             0,                       // td0[15:0]=0 (td0 = 1<<20)
             16,                      // td0[31:16]=16 | td1[15:0]=0 (td1 = 1<<20)
             16 | (KSTEP << 16),      // td1[31:16]=16 | tile_dim0=32
             TILE,                    // tile_dim1=128, tile_dim2=0
             D_SZ,                    // tensor_dim0_stride = 1024
             0, 0 };
  return g1;
}

// ---------------------------------------------------------- TDM GEMM kernel
__global__ __launch_bounds__(256)
void nt_gemm_tdm(const unsigned short* __restrict__ Lhi, const unsigned short* __restrict__ Llo,
                 const unsigned short* __restrict__ Rhi, const unsigned short* __restrict__ Rlo,
                 const float* __restrict__ lnArr, const float* __restrict__ rnArr,
                 float* __restrict__ rowsum, float* __restrict__ diag) {
  const int bj = blockIdx.x;
  const int bi = blockIdx.y;
  if (bj < bi) return;                 // upper triangle only
  const bool diagTile = (bi == bj);

  __shared__ unsigned short sT[2][4][TILE * LDST];  // [buf][Ahi,Alo,Bhi,Blo] 80KB
  __shared__ float lnS[TILE], rnS[TILE];

  const int tid  = threadIdx.x;
  const int wave = tid >> 5;
  const int lane = tid & 31;
  const int half = lane >> 4;
  const int l16  = lane & 15;
  const int wrBase = (wave >> 1) * 32;
  const int wcBase = (wave & 1) * 64;
  const int biBase = bi * TILE;
  const int bjBase = bj * TILE;

  if (tid < TILE) {
    lnS[tid] = lnArr[biBase + tid];
    rnS[tid] = rnArr[bjBase + tid];
  }

  v8f acc[2][4];
  for (int sr = 0; sr < 2; ++sr)
    for (int sc = 0; sc < 4; ++sc)
      acc[sr][sc] = (v8f)0.f;

  const v8i g1 = tdm_group1();
  const long long aRow = (long long)biBase * D_SZ;
  const long long bRow = (long long)bjBase * D_SZ;

  // prologue: DMA buffer 0
  if (wave == 0) {
    tdm_tile_load(Lhi, aRow, &sT[0][0][0], g1);
    tdm_tile_load(Llo, aRow, &sT[0][1][0], g1);
    tdm_tile_load(Rhi, bRow, &sT[0][2][0], g1);
    tdm_tile_load(Rlo, bRow, &sT[0][3][0], g1);
    __builtin_amdgcn_s_wait_tensorcnt(0);
  }
  __syncthreads();

  int buf = 0;
  for (int k0 = 0; k0 < D_SZ; k0 += KSTEP) {
    const bool more = (k0 + KSTEP) < D_SZ;
    if (more && wave == 0) {           // DMA next buffer, overlapped with WMMA
      const int nb = buf ^ 1;
      tdm_tile_load(Lhi, aRow + k0 + KSTEP, &sT[nb][0][0], g1);
      tdm_tile_load(Llo, aRow + k0 + KSTEP, &sT[nb][1][0], g1);
      tdm_tile_load(Rhi, bRow + k0 + KSTEP, &sT[nb][2][0], g1);
      tdm_tile_load(Rlo, bRow + k0 + KSTEP, &sT[nb][3][0], g1);
    }
    compute_step(&sT[buf][0][0], &sT[buf][1][0], &sT[buf][2][0], &sT[buf][3][0],
                 acc, wrBase, wcBase, l16, half);
    if (more) {
      if (wave == 0) __builtin_amdgcn_s_wait_tensorcnt(0);
      __syncthreads();                 // publish next buffer / retire current reads
    }
    buf ^= 1;
  }

  nt_epilogue(acc, lnS, rnS, rowsum, diag, diagTile, biBase, bjBase,
              wrBase, wcBase, l16, half);
}
#endif // HAVE_TDM

// --------------------------------------------- prep: split + norms + init
__global__ void nt_prep(const float* __restrict__ L, const float* __restrict__ R,
                        unsigned short* __restrict__ Lhi, unsigned short* __restrict__ Llo,
                        unsigned short* __restrict__ Rhi, unsigned short* __restrict__ Rlo,
                        float* __restrict__ ln, float* __restrict__ rn,
                        float* __restrict__ rowsum) {
  const int row = blockIdx.x;
  const int t = threadIdx.x;                       // 256 threads = 256 float4 per row
  const float* src = blockIdx.y ? R : L;
  unsigned short* hiA = blockIdx.y ? Rhi : Lhi;
  unsigned short* loA = blockIdx.y ? Rlo : Llo;
  const size_t base = (size_t)row * D_SZ + (size_t)t * 4;

  float4 v = *(const float4*)(&src[base]);
  unsigned short h0 = f32_to_bf16_rne(v.x), h1 = f32_to_bf16_rne(v.y);
  unsigned short h2 = f32_to_bf16_rne(v.z), h3 = f32_to_bf16_rne(v.w);
  unsigned short l0 = f32_to_bf16_rne(v.x - bf16u_to_f32(h0));
  unsigned short l1 = f32_to_bf16_rne(v.y - bf16u_to_f32(h1));
  unsigned short l2 = f32_to_bf16_rne(v.z - bf16u_to_f32(h2));
  unsigned short l3 = f32_to_bf16_rne(v.w - bf16u_to_f32(h3));
  *(uint2*)(&hiA[base]) = make_uint2((unsigned)h0 | ((unsigned)h1 << 16),
                                     (unsigned)h2 | ((unsigned)h3 << 16));
  *(uint2*)(&loA[base]) = make_uint2((unsigned)l0 | ((unsigned)l1 << 16),
                                     (unsigned)l2 | ((unsigned)l3 << 16));

  float s = v.x * v.x + v.y * v.y + v.z * v.z + v.w * v.w;
  __shared__ float red[256];
  red[t] = s;
  __syncthreads();
  for (int o = 128; o > 0; o >>= 1) {
    if (t < o) red[t] += red[t + o];
    __syncthreads();
  }
  if (t == 0) {
    (blockIdx.y ? rn : ln)[row] = sqrtf(red[0]);
    if (blockIdx.y == 0) rowsum[row] = 0.f;        // accumulator init (ws is poisoned)
  }
}

// ---------------------------------------------------------- fallback kernels
__global__ void nt_norms(const float* __restrict__ L, const float* __restrict__ R,
                         float* __restrict__ ln, float* __restrict__ rn,
                         float* __restrict__ rowsum) {
  const int row = blockIdx.x;
  const float* src = blockIdx.y ? R : L;
  float s = 0.f;
  for (int c = threadIdx.x; c < D_SZ; c += 256) {
    float v = src[(size_t)row * D_SZ + c];
    s += v * v;
  }
  __shared__ float red[256];
  red[threadIdx.x] = s;
  __syncthreads();
  for (int o = 128; o > 0; o >>= 1) {
    if (threadIdx.x < o) red[threadIdx.x] += red[threadIdx.x + o];
    __syncthreads();
  }
  if (threadIdx.x == 0) {
    (blockIdx.y ? rn : ln)[row] = sqrtf(red[0]);
    if (blockIdx.y == 0) rowsum[row] = 0.f;
  }
}

__global__ __launch_bounds__(256)
void nt_gemm(const float* __restrict__ L, const float* __restrict__ R,
             const float* __restrict__ lnArr, const float* __restrict__ rnArr,
             float* __restrict__ rowsum, float* __restrict__ diag) {
  const int bj = blockIdx.x;
  const int bi = blockIdx.y;
  if (bj < bi) return;
  const bool diagTile = (bi == bj);

  __shared__ unsigned short sAhi[TILE * LDST];
  __shared__ unsigned short sAlo[TILE * LDST];
  __shared__ unsigned short sBhi[TILE * LDST];
  __shared__ unsigned short sBlo[TILE * LDST];
  __shared__ float lnS[TILE], rnS[TILE];

  const int tid  = threadIdx.x;
  const int wave = tid >> 5;
  const int lane = tid & 31;
  const int half = lane >> 4;
  const int l16  = lane & 15;
  const int wrBase = (wave >> 1) * 32;
  const int wcBase = (wave & 1) * 64;
  const int biBase = bi * TILE;
  const int bjBase = bj * TILE;

  if (tid < TILE) {
    lnS[tid] = lnArr[biBase + tid];
    rnS[tid] = rnArr[bjBase + tid];
  }

  v8f acc[2][4];
  for (int sr = 0; sr < 2; ++sr)
    for (int sc = 0; sc < 4; ++sc)
      acc[sr][sc] = (v8f)0.f;

  for (int k0 = 0; k0 < D_SZ; k0 += KSTEP) {
    if (k0) __syncthreads();
    for (int p = 0; p < 4; ++p) {
      const int q  = tid + 256 * p;
      const int r  = q >> 3;
      const int c4 = q & 7;
      {
        float4 av = *(const float4*)(&L[(size_t)(biBase + r) * D_SZ + k0 + c4 * 4]);
        unsigned short h0 = f32_to_bf16_rne(av.x), h1 = f32_to_bf16_rne(av.y);
        unsigned short h2 = f32_to_bf16_rne(av.z), h3 = f32_to_bf16_rne(av.w);
        unsigned short l0 = f32_to_bf16_rne(av.x - bf16u_to_f32(h0));
        unsigned short l1 = f32_to_bf16_rne(av.y - bf16u_to_f32(h1));
        unsigned short l2 = f32_to_bf16_rne(av.z - bf16u_to_f32(h2));
        unsigned short l3 = f32_to_bf16_rne(av.w - bf16u_to_f32(h3));
        *(uint2*)(&sAhi[r * LDST + c4 * 4]) =
            make_uint2((unsigned)h0 | ((unsigned)h1 << 16), (unsigned)h2 | ((unsigned)h3 << 16));
        *(uint2*)(&sAlo[r * LDST + c4 * 4]) =
            make_uint2((unsigned)l0 | ((unsigned)l1 << 16), (unsigned)l2 | ((unsigned)l3 << 16));
      }
      {
        float4 bv = *(const float4*)(&R[(size_t)(bjBase + r) * D_SZ + k0 + c4 * 4]);
        unsigned short h0 = f32_to_bf16_rne(bv.x), h1 = f32_to_bf16_rne(bv.y);
        unsigned short h2 = f32_to_bf16_rne(bv.z), h3 = f32_to_bf16_rne(bv.w);
        unsigned short l0 = f32_to_bf16_rne(bv.x - bf16u_to_f32(h0));
        unsigned short l1 = f32_to_bf16_rne(bv.y - bf16u_to_f32(h1));
        unsigned short l2 = f32_to_bf16_rne(bv.z - bf16u_to_f32(h2));
        unsigned short l3 = f32_to_bf16_rne(bv.w - bf16u_to_f32(h3));
        *(uint2*)(&sBhi[r * LDST + c4 * 4]) =
            make_uint2((unsigned)h0 | ((unsigned)h1 << 16), (unsigned)h2 | ((unsigned)h3 << 16));
        *(uint2*)(&sBlo[r * LDST + c4 * 4]) =
            make_uint2((unsigned)l0 | ((unsigned)l1 << 16), (unsigned)l2 | ((unsigned)l3 << 16));
      }
    }
    __syncthreads();
    compute_step(sAhi, sAlo, sBhi, sBlo, acc, wrBase, wcBase, l16, half);
  }

  nt_epilogue(acc, lnS, rnS, rowsum, diag, diagTile, biBase, bjBase,
              wrBase, wcBase, l16, half);
}

// ---------------------------------------------------------------- kernel 3
__global__ void nt_finalize(const float* __restrict__ rowsum,
                            const float* __restrict__ diag,
                            float* __restrict__ out) {
  float s = 0.f;
  for (int i = threadIdx.x; i < B_SZ; i += 256)
    s += __logf(rowsum[i]) - diag[i];
  __shared__ float red[256];
  red[threadIdx.x] = s;
  __syncthreads();
  for (int o = 128; o > 0; o >>= 1) {
    if (threadIdx.x < o) red[threadIdx.x] += red[threadIdx.x + o];
    __syncthreads();
  }
  if (threadIdx.x == 0) out[0] = red[0] * (1.0f / (float)B_SZ);
}

// ---------------------------------------------------------------- launcher
extern "C" void kernel_launch(void* const* d_in, const int* in_sizes, int n_in,
                              void* d_out, int out_size, void* d_ws, size_t ws_size,
                              hipStream_t stream) {
  (void)in_sizes; (void)n_in; (void)out_size;
  const float* L = (const float*)d_in[0];
  const float* R = (const float*)d_in[1];
  float* out = (float*)d_out;
  float* ws  = (float*)d_ws;
  float* ln     = ws;
  float* rn     = ws + B_SZ;
  float* rowsum = ws + 2 * B_SZ;
  float* diag   = ws + 3 * B_SZ;

  const size_t N = (size_t)B_SZ * D_SZ;
  const size_t need = 4 * B_SZ * sizeof(float) + 4 * N * sizeof(unsigned short);

  dim3 gG(B_SZ / TILE, B_SZ / TILE);   // 32x32, lower triangle early-exits

#ifdef HAVE_TDM
  if (ws_size >= need) {
    unsigned short* sp  = (unsigned short*)(ws + 4 * B_SZ);
    unsigned short* Lhi = sp;
    unsigned short* Llo = sp + N;
    unsigned short* Rhi = sp + 2 * N;
    unsigned short* Rlo = sp + 3 * N;

    dim3 gP(B_SZ, 2);
    nt_prep<<<gP, 256, 0, stream>>>(L, R, Lhi, Llo, Rhi, Rlo, ln, rn, rowsum);
    nt_gemm_tdm<<<gG, 256, 0, stream>>>(Lhi, Llo, Rhi, Rlo, ln, rn, rowsum, diag);
    nt_finalize<<<1, 256, 0, stream>>>(rowsum, diag, out);
    return;
  }
#endif
  (void)need;
  dim3 gN(B_SZ, 2);
  nt_norms<<<gN, 256, 0, stream>>>(L, R, ln, rn, rowsum);
  nt_gemm<<<gG, 256, 0, stream>>>(L, R, ln, rn, rowsum, diag);
  nt_finalize<<<1, 256, 0, stream>>>(rowsum, diag, out);
}
